// E3Conv_66881230733948
// MI455X (gfx1250) — matrix operand order repaired and checked by reading, not compiled, not run on hardware
//
#include <hip/hip_runtime.h>
#include <hip/hip_bf16.h>

#define NN_NODES 10000
#define NN_EDGES 131072
#define W4_COLS 1792   // 8*8*(16+8+4)
#define SQ3 1.7320508075688772f
#define SQ5 2.23606797749979f
#define SQ15 3.872983346207417f

typedef __attribute__((ext_vector_type(16))) _Float16 v16h;
typedef __attribute__((ext_vector_type(8)))  float    v8f;

// workspace layout (bytes)
#define WS_W3   ((size_t)0)                     // E*64 f16  = 16777216
#define WS_SH   ((size_t)16777216)              // E*8  f32  = 4194304
#define WS_AI   ((size_t)(WS_SH + 4194304))     // N*8  f32  (padded to 327680)
#define WS_W4   ((size_t)(WS_AI + 327680))      // 112*2*512 f16 = 229376
#define WS_ACC  ((size_t)(WS_W4 + 229376))      // N*60 f32  = 2400000
#define WS_CNT  ((size_t)(WS_ACC + 2400000))    // N    f32  = 40000
#define WS_W2S  ((size_t)(WS_CNT + 40000))      // 4*2*512 f16 = 8192
#define WS_W3S  ((size_t)(WS_W2S + 8192))       // 4*2*512 f16 = 8192

// fast reciprocal: single v_rcp_f32 (TRANS op, co-issues with WMMA) instead of
// the ~10-instruction IEEE division chain.
__device__ __forceinline__ float frcp(float x) { return __builtin_amdgcn_rcpf(x); }
__device__ __forceinline__ float silu(float x) { return x * frcp(1.0f + __expf(-x)); }

// ---------------- zero accumulators ----------------
__global__ __launch_bounds__(256) void k_zero(float* p, int n) {
  int i = blockIdx.x * 256 + threadIdx.x;
  if (i < n) p[i] = 0.0f;
}

// ---- pack an f32 (64 x ncols) weight matrix into WMMA B-fragments (f16, scaled) ----
// fragment fi = t*2 + kk (t = 16-col tile, kk = K-half). 512 halfs per fragment.
// lane L holds 16 halfs: n = L%16 (col c = 16t+n), K = kk*32 + (j>=8?16:0) + (L/16)*8 + (j&7).
__global__ __launch_bounds__(256) void k_swizzle(const float* __restrict__ src,
                                                 _Float16* __restrict__ dst,
                                                 int ncols, int nelems, float scale) {
  int i = blockIdx.x * 256 + threadIdx.x;
  if (i >= nelems) return;
  int fi = i >> 9, r = i & 511;
  int lane = r >> 4, j = r & 15;
  int t = fi >> 1, kk = fi & 1;
  int k = kk * 32 + ((j & 8) ? 16 : 0) + (lane >> 4) * 8 + (j & 7);
  int c = t * 16 + (lane & 15);
  dst[i] = (_Float16)(src[k * ncols + c] * scale);
}

// ---------------- node MLP: embed -> 64 -> 32 -> 8 (VALU, tiny) ----------------
__global__ __launch_bounds__(256) void k_node_mlp(
    const int* __restrict__ A, const float* __restrict__ embed,
    const float* __restrict__ W1, const float* __restrict__ b1,
    const float* __restrict__ W2, const float* __restrict__ b2,
    const float* __restrict__ W3, const float* __restrict__ b3,
    float* __restrict__ aibuf) {
  __shared__ float sW1[16 * 64], sB1[64], sW2[64 * 32], sB2[32], sW3[32 * 8], sB3[8];
  int tid = threadIdx.x;
  for (int i = tid; i < 16 * 64; i += 256) sW1[i] = W1[i];
  for (int i = tid; i < 64 * 32; i += 256) sW2[i] = W2[i];
  for (int i = tid; i < 32 * 8;  i += 256) sW3[i] = W3[i];
  if (tid < 64) sB1[tid] = b1[tid];
  if (tid < 32) sB2[tid] = b2[tid];
  if (tid < 8)  sB3[tid] = b3[tid];
  __syncthreads();
  int n = blockIdx.x * 256 + tid;
  if (n >= NN_NODES) return;
  float h0[16];
  int a = A[n];
  #pragma unroll
  for (int k = 0; k < 16; ++k) h0[k] = embed[a * 16 + k];
  float h1[64];
  #pragma unroll
  for (int j = 0; j < 64; ++j) {
    float s = sB1[j];
    #pragma unroll
    for (int k = 0; k < 16; ++k) s += h0[k] * sW1[k * 64 + j];
    h1[j] = silu(s);
  }
  float h2[32];
  #pragma unroll
  for (int j = 0; j < 32; ++j) {
    float s = sB2[j];
    #pragma unroll
    for (int k = 0; k < 64; ++k) s += h1[k] * sW2[k * 32 + j];
    h2[j] = silu(s);
  }
  #pragma unroll
  for (int j = 0; j < 8; ++j) {
    float s = sB3[j];
    #pragma unroll
    for (int k = 0; k < 32; ++k) s += h2[k] * sW3[k * 8 + j];
    aibuf[n * 8 + j] = s;
  }
}

// ------- edge geometry + SH + radial + layer1 (VALU) + layers 2,3 via WMMA -------
// Block = 256 threads = 8 waves; block owns 128 edges, each wave a 16-edge tile.
__global__ __launch_bounds__(256) void k_edge_mlp(
    const float* __restrict__ pos, const int* __restrict__ batch,
    const int* __restrict__ esrc, const int* __restrict__ edst,
    const float* __restrict__ shifts, const float* __restrict__ cell,
    const float* __restrict__ fcW1,
    const _Float16* __restrict__ w2swz, const _Float16* __restrict__ w3swz,
    _Float16* __restrict__ w3buf, float* __restrict__ shbuf,
    float* __restrict__ counts) {
  __shared__ float sW1[10 * 64];
  __shared__ _Float16 hb1[128 * 64];   // 16 KB
  __shared__ _Float16 hb2[128 * 64];   // 16 KB
  const int tid = threadIdx.x;
  for (int i = tid; i < 10 * 64; i += 256) sW1[i] = fcW1[i];
  __syncthreads();

  // ---- layer 1: each thread computes half of h1 for one edge ----
  const int eloc = tid >> 1;          // 0..127
  const int jh   = (tid & 1) * 32;    // which half of h1
  const int e = blockIdx.x * 128 + eloc;
  {
    int s = esrc[e], d = edst[e];
    int b = batch[s];
    const float* C = cell + (size_t)b * 9;
    float sx = shifts[e * 3 + 0], sy = shifts[e * 3 + 1], sz = shifts[e * 3 + 2];
    float vx = pos[d * 3 + 0] - pos[s * 3 + 0] + sx * C[0] + sy * C[3] + sz * C[6];
    float vy = pos[d * 3 + 1] - pos[s * 3 + 1] + sx * C[1] + sy * C[4] + sz * C[7];
    float vz = pos[d * 3 + 2] - pos[s * 3 + 2] + sx * C[2] + sy * C[5] + sz * C[8];
    float len = sqrtf(vx * vx + vy * vy + vz * vz);
    float inv = frcp(fmaxf(len, 1e-9f));
    float x = vx * inv, y = vy * inv, z = vz * inv;
    if ((tid & 1) == 0) {
      shbuf[(size_t)e * 8 + 0] = SQ3 * x;
      shbuf[(size_t)e * 8 + 1] = SQ3 * y;
      shbuf[(size_t)e * 8 + 2] = SQ3 * z;
      shbuf[(size_t)e * 8 + 3] = SQ15 * x * z;
      shbuf[(size_t)e * 8 + 4] = SQ15 * x * y;
      shbuf[(size_t)e * 8 + 5] = SQ5 * (y * y - 0.5f * (x * x + z * z));
      shbuf[(size_t)e * 8 + 6] = SQ15 * y * z;
      shbuf[(size_t)e * 8 + 7] = 0.5f * SQ15 * (z * z - x * x);
      unsafeAtomicAdd(&counts[d], 1.0f);
    }
    // soft one-hot (sqrt(10) scale cancels layer-1's /sqrt(10))
    float emb[10];
    const float step = 4.0f / 11.0f;
    #pragma unroll
    for (int k = 0; k < 10; ++k) {
      float dd = (len - step * (float)(k + 1)) / step;
      emb[k] = __expf(-dd * dd) * (1.0f / 1.12f);
    }
    #pragma unroll
    for (int jj = 0; jj < 32; ++jj) {
      int j = jh + jj;
      float acc = 0.0f;
      #pragma unroll
      for (int k = 0; k < 10; ++k) acc += emb[k] * sW1[k * 64 + j];
      hb1[eloc * 64 + j] = (_Float16)silu(acc);
    }
  }
  __syncthreads();

  // ---- layers 2 & 3: per-wave (16x64)@(64x64) WMMA, SiLU, LDS round-trip ----
  const int lane = tid & 31, wave = tid >> 5;
  const int mA = lane & 15, halfA = lane >> 4;   // A-fragment row / K-half
  const int nCol = lane & 15, halfD = lane >> 4; // D layout: n, row-half
  const int rowBase = wave * 16;

  v16h bf2[8], bf3[8];
  {
    const v16h* p2 = (const v16h*)w2swz;
    const v16h* p3 = (const v16h*)w3swz;
    #pragma unroll
    for (int fi = 0; fi < 8; ++fi) { bf2[fi] = p2[fi * 32 + lane]; bf3[fi] = p3[fi * 32 + lane]; }
  }

  // layer 2: hb1 -> hb2
  {
    v16h a0, a1;
    const _Float16* rowp = hb1 + (rowBase + mA) * 64;
    #pragma unroll
    for (int j = 0; j < 16; ++j) {
      int k = ((j & 8) ? 16 : 0) + halfA * 8 + (((j >> 1) & 3) << 1) + (j & 1);
      a0[j] = rowp[k];
      a1[j] = rowp[k + 32];
    }
    #pragma unroll
    for (int nt = 0; nt < 4; ++nt) {
      v8f acc = {};
      acc = __builtin_amdgcn_wmma_f32_16x16x32_f16(false, a0, false, bf2[nt * 2 + 0], (short)0, acc, false, false);
      acc = __builtin_amdgcn_wmma_f32_16x16x32_f16(false, a1, false, bf2[nt * 2 + 1], (short)0, acc, false, false);
      #pragma unroll
      for (int r = 0; r < 8; ++r) {
        int m = r + (halfD << 3);
        hb2[(rowBase + m) * 64 + nt * 16 + nCol] = (_Float16)silu(acc[r]);
      }
    }
  }
  __syncthreads();

  // layer 3: hb2 -> hb1
  {
    v16h a0, a1;
    const _Float16* rowp = hb2 + (rowBase + mA) * 64;
    #pragma unroll
    for (int j = 0; j < 16; ++j) {
      int k = ((j & 8) ? 16 : 0) + halfA * 8 + (((j >> 1) & 3) << 1) + (j & 1);
      a0[j] = rowp[k];
      a1[j] = rowp[k + 32];
    }
    #pragma unroll
    for (int nt = 0; nt < 4; ++nt) {
      v8f acc = {};
      acc = __builtin_amdgcn_wmma_f32_16x16x32_f16(false, a0, false, bf3[nt * 2 + 0], (short)0, acc, false, false);
      acc = __builtin_amdgcn_wmma_f32_16x16x32_f16(false, a1, false, bf3[nt * 2 + 1], (short)0, acc, false, false);
      #pragma unroll
      for (int r = 0; r < 8; ++r) {
        int m = r + (halfD << 3);
        hb1[(rowBase + m) * 64 + nt * 16 + nCol] = (_Float16)silu(acc[r]);
      }
    }
  }
  __syncthreads();

  // ---- vectorized copy of the 128x64 f16 tile to global w3buf ----
  {
    const uint4* src = (const uint4*)hb1;                 // 1024 uint4
    uint4* dst = (uint4*)(w3buf + (size_t)blockIdx.x * 128 * 64);
    #pragma unroll
    for (int it = 0; it < 4; ++it) dst[it * 256 + tid] = src[it * 256 + tid];
  }
}

// ------- fused: (16 edges x 64) @ (64 x 1792) via WMMA f16 -> f32, then contract -------
__global__ __launch_bounds__(256) void k_gemm_contract(
    const _Float16* __restrict__ w3buf, const _Float16* __restrict__ w4buf,
    const float* __restrict__ aibuf, const float* __restrict__ shbuf,
    const int* __restrict__ esrc, const int* __restrict__ edst,
    float* __restrict__ accum) {
  __shared__ float AiS[16][8], AiD[16][8], shT[16][8];
  __shared__ float sS[16][28];
  __shared__ int dstT[16];
  const int tid = threadIdx.x;
  const int tile = blockIdx.x;
  if (tid < 16) dstT[tid] = edst[tile * 16 + tid];
  if (tid < 128) {
    int e = tid >> 3, c = tid & 7;
    int sn = esrc[tile * 16 + e];
    AiS[e][c] = aibuf[sn * 8 + c];
    shT[e][c] = shbuf[(size_t)(tile * 16 + e) * 8 + c];
  } else {
    int t2 = tid - 128;
    int e = t2 >> 3, c = t2 & 7;
    int dn = edst[tile * 16 + e];
    AiD[e][c] = aibuf[dn * 8 + c];
  }
  for (int i = tid; i < 16 * 28; i += 256) ((float*)sS)[i] = 0.0f;
  __syncthreads();

  const int lane = tid & 31, wave = tid >> 5;
  const int mA = lane & 15, halfA = lane >> 4;
  // Build A fragments (16x32 f16 per WMMA, K split 0..31 / 32..63)
  v16h a0, a1;
  const _Float16* wrow = w3buf + (size_t)(tile * 16 + mA) * 64;
  #pragma unroll
  for (int j = 0; j < 16; ++j) {
    int k = ((j & 8) ? 16 : 0) + halfA * 8 + (((j >> 1) & 3) << 1) + (j & 1);
    a0[j] = wrow[k];
    a1[j] = wrow[k + 32];
  }
  const int nCol = lane & 15, halfD = lane >> 4;
  // each wave owns 14 of the 112 column tiles
  for (int tt = 0; tt < 14; ++tt) {
    int t = wave * 14 + tt;
    const v16h* bp = (const v16h*)(w4buf + (size_t)t * 1024);
    if (tt < 13) __builtin_prefetch(w4buf + (size_t)(t + 1) * 1024 + lane * 32, 0, 1);
    v16h b0 = bp[lane];
    v16h b1 = bp[32 + lane];
    v8f acc = {};
    acc = __builtin_amdgcn_wmma_f32_16x16x32_f16(false, a0, false, b0, (short)0, acc, false, false);
    acc = __builtin_amdgcn_wmma_f32_16x16x32_f16(false, a1, false, b1, (short)0, acc, false, false);
    // column c -> (u, v, w-slot) of the e3nn tensor product weights
    int c = t * 16 + nCol;
    int u, v, sidx;
    if (c < 1024)      { u = c >> 7;              v = (c >> 4) & 7; sidx = c & 15; }
    else if (c < 1536) { int g = c - 1024; u = g >> 6; v = (g >> 3) & 7; sidx = 16 + (g & 7); }
    else               { int g = c - 1536; u = g >> 5; v = (g >> 2) & 7; sidx = 24 + (g & 3); }
    #pragma unroll
    for (int r = 0; r < 8; ++r) {
      int m = r + (halfD << 3);
      float val = acc[r] * AiS[m][u] * AiD[m][v];
      atomicAdd(&sS[m][sidx], val);
    }
  }
  __syncthreads();
  // features = s * sh * (1/MUL), scatter-add by edge_dst
  for (int i = tid; i < 16 * 60; i += 256) {
    int e = i / 60, f = i % 60;
    float val;
    if (f < 16) {
      val = sS[e][f];
    } else if (f < 40) {
      int g = f - 16;
      val = sS[e][16 + g / 3] * shT[e][g % 3];
    } else {
      int g = f - 40;
      val = sS[e][24 + g / 5] * shT[e][3 + g % 5];
    }
    val *= 0.125f;
    unsafeAtomicAdd(&accum[(size_t)dstT[e] * 60 + f], val);
  }
}

// ---------------- out = accum / max(counts, 1) ----------------
__global__ __launch_bounds__(256) void k_finalize(const float* __restrict__ accum,
                                                  const float* __restrict__ counts,
                                                  float* __restrict__ out) {
  int i = blockIdx.x * 256 + threadIdx.x;
  if (i >= NN_NODES * 60) return;
  int node = i / 60;
  out[i] = accum[i] * frcp(fmaxf(counts[node], 1.0f));
}

extern "C" void kernel_launch(void* const* d_in, const int* in_sizes, int n_in,
                              void* d_out, int out_size, void* d_ws, size_t ws_size,
                              hipStream_t stream) {
  const float* pos     = (const float*)d_in[0];
  const int*   A       = (const int*)d_in[1];
  const int*   batch   = (const int*)d_in[2];
  const int*   esrc    = (const int*)d_in[3];
  const int*   edst    = (const int*)d_in[4];
  const float* shifts  = (const float*)d_in[5];
  const float* cell    = (const float*)d_in[6];
  const float* embed   = (const float*)d_in[7];
  const float* fitW1   = (const float*)d_in[8];
  const float* fitb1   = (const float*)d_in[9];
  const float* fitW2   = (const float*)d_in[10];
  const float* fitb2   = (const float*)d_in[11];
  const float* fitW3   = (const float*)d_in[12];
  const float* fitb3   = (const float*)d_in[13];
  const float* fcW1    = (const float*)d_in[14];
  const float* fcW2    = (const float*)d_in[15];
  const float* fcW3    = (const float*)d_in[16];
  const float* fcW4    = (const float*)d_in[17];

  char* ws = (char*)d_ws;
  _Float16* w3buf = (_Float16*)(ws + WS_W3);
  float*    shbuf = (float*)(ws + WS_SH);
  float*    aibuf = (float*)(ws + WS_AI);
  _Float16* w4buf = (_Float16*)(ws + WS_W4);
  float*    accum = (float*)(ws + WS_ACC);
  float*    counts= (float*)(ws + WS_CNT);
  _Float16* w2swz = (_Float16*)(ws + WS_W2S);
  _Float16* w3swz = (_Float16*)(ws + WS_W3S);
  float*    out   = (float*)d_out;

  // accum (600000) and counts (10000) are contiguous: zero both in one pass
  int nz = NN_NODES * 60 + NN_NODES;
  k_zero<<<(nz + 255) / 256, 256, 0, stream>>>(accum, nz);
  // pack B-fragment weight buffers (scale folds the pre-activation /8)
  k_swizzle<<<(W4_COLS * 64 + 255) / 256, 256, 0, stream>>>(fcW4, w4buf, W4_COLS, W4_COLS * 64, 0.125f);
  k_swizzle<<<(64 * 64 + 255) / 256, 256, 0, stream>>>(fcW2, w2swz, 64, 64 * 64, 0.125f);
  k_swizzle<<<(64 * 64 + 255) / 256, 256, 0, stream>>>(fcW3, w3swz, 64, 64 * 64, 0.125f);
  k_node_mlp<<<(NN_NODES + 255) / 256, 256, 0, stream>>>(
      A, embed, fitW1, fitb1, fitW2, fitb2, fitW3, fitb3, aibuf);
  k_edge_mlp<<<NN_EDGES / 128, 256, 0, stream>>>(
      pos, batch, esrc, edst, shifts, cell, fcW1, w2swz, w3swz, w3buf, shbuf, counts);
  k_gemm_contract<<<NN_EDGES / 16, 256, 0, stream>>>(
      w3buf, w4buf, aibuf, shbuf, esrc, edst, accum);
  k_finalize<<<(NN_NODES * 60 + 255) / 256, 256, 0, stream>>>(accum, counts, out);
}
